// CustomHBLSTMCell_76416058130986
// MI455X (gfx1250) — matrix operand order repaired and checked by reading, compiled, not run on hardware
//
#include <hip/hip_runtime.h>

#define B_DIM  65536
#define IN_DIM 256
#define H_DIM  256
#define NCOL   1280   // 5 * H
#define KDIM   512    // IN + H

typedef __attribute__((ext_vector_type(16))) __bf16        v16bf;
typedef __attribute__((ext_vector_type(8)))  float         v8f;
typedef __attribute__((ext_vector_type(4)))  unsigned int  v4u;

union FragBF {
  v16bf v;
  v4u   u[2];
};

__device__ __forceinline__ unsigned short f32_to_bf16(float f) {
  unsigned int u = __float_as_uint(f);
  u += 0x7FFFu + ((u >> 16) & 1u);   // round-to-nearest-even
  return (unsigned short)(u >> 16);
}

// Pack [Wx | Uh] (fp32) into bf16, K-major: Wpack[n*512 + k], n = g*256 + h.
__global__ void __launch_bounds__(256) pack_weights(
    const float* __restrict__ Wx, const float* __restrict__ Uh,
    unsigned short* __restrict__ Wpack) {
  int e = blockIdx.x * 256 + threadIdx.x;      // 0 .. 1280*512-1
  int n = e >> 9;
  int k = e & 511;
  float v = (k < IN_DIM) ? Wx[n * IN_DIM + k] : Uh[n * H_DIM + (k - IN_DIM)];
  Wpack[e] = f32_to_bf16(v);
}

__global__ void __launch_bounds__(256) hblstm_kernel(
    const float* __restrict__ x,       const float* __restrict__ h_prev,
    const float* __restrict__ k_prev,  const float* __restrict__ kp_prev,
    const float* __restrict__ bx,      const float* __restrict__ bh,
    const unsigned short* __restrict__ Wpack,
    float* __restrict__ out) {
  __shared__ __align__(16) unsigned short aLds[16 * KDIM];   // 16 KB

  const int tid   = threadIdx.x;
  const int lane  = tid & 31;
  const int wave  = tid >> 5;
  const int mBase = blockIdx.x * 16;

  // Stage A tile (16 rows of [x | h_prev]) fp32 -> bf16 into LDS, coalesced.
  for (int e = tid; e < 16 * KDIM; e += 256) {
    int r = e >> 9;
    int k = e & 511;
    float v = (k < IN_DIM) ? x[(mBase + r) * IN_DIM + k]
                           : h_prev[(mBase + r) * H_DIM + (k - IN_DIM)];
    aLds[e] = f32_to_bf16(v);
  }
  __syncthreads();

  const int hiHalf = (lane >> 4) & 1;
  // Each wave owns two 16-wide h-tiles: full H=256 covered by 8 waves x 2.
  int hCol[2];
  hCol[0] = wave * 16 + (lane & 15);
  hCol[1] = 128 + wave * 16 + (lane & 15);

  v8f acc[2][5] = {};   // [h-tile][gate f,i,o,u,k] f32 accumulators

  const unsigned short* aRow = &aLds[(lane & 15) * KDIM];

  for (int kt = 0; kt < 16; ++kt) {
    // A fragment per ISA 16-bit A layout:
    //   lanes 0-15:  K = kt*32 + {0..7} and {16..23}
    //   lanes 16-31: K = kt*32 + {8..15} and {24..31}
    const int koffA = kt * 32 + (hiHalf ? 8 : 0);
    FragBF a;
    a.u[0] = *(const v4u*)(aRow + koffA);
    a.u[1] = *(const v4u*)(aRow + koffA + 16);

    // B fragment: lane n holds column hCol[t], same K order as A:
    //   lanes 0-15: K = kt*32 + {0..7} and {16..23}, lanes 16-31: {8..15} and {24..31}.
    const int koffB = kt * 32 + (hiHalf ? 8 : 0);
#pragma unroll
    for (int t = 0; t < 2; ++t) {
#pragma unroll
      for (int g = 0; g < 5; ++g) {
        const unsigned short* bRow =
            Wpack + (size_t)(g * H_DIM + hCol[t]) * KDIM + koffB;
        FragBF b;
        b.u[0] = *(const v4u*)(bRow);
        b.u[1] = *(const v4u*)(bRow + 16);
        acc[t][g] = __builtin_amdgcn_wmma_f32_16x16x32_bf16(
            false, a.v, false, b.v, (short)0, acc[t][g], false, false);
      }
    }
  }

  const int BH = B_DIM * H_DIM;
#pragma unroll
  for (int t = 0; t < 2; ++t) {
    // Combined bias per gate for this lane's column (uniform over j).
    float bias[5];
#pragma unroll
    for (int g = 0; g < 5; ++g)
      bias[g] = bx[g * H_DIM + hCol[t]] + bh[g * H_DIM + hCol[t]];

    // Epilogue: C/D 16x16 f32 layout — VGPR j of this lane is
    //   row M = j + (hiHalf ? 8 : 0), col N = lane & 15.
#pragma unroll
    for (int j = 0; j < 8; ++j) {
      int row = mBase + j + (hiHalf ? 8 : 0);
      int idx = row * H_DIM + hCol[t];

      float pf = acc[t][0][j] + bias[0];
      float pi = acc[t][1][j] + bias[1];
      float po = acc[t][2][j] + bias[2];
      float pu = acc[t][3][j] + bias[3];
      float pk = acc[t][4][j] + bias[4];

      float fg = 1.0f / (1.0f + __expf(-pf));
      float ig = 1.0f / (1.0f + __expf(-pi));
      float og = 1.0f / (1.0f + __expf(-po));
      float ug = 1.0f / (1.0f + __expf(-pu));

      float kp = ug * tanhf(pk) + (1.0f - ug) * kp_prev[idx];
      float ks = fg * k_prev[idx] + ig * kp;
      float hs = og * tanhf(ks);

      out[idx]          = hs;
      out[BH + idx]     = ks;
      out[2 * BH + idx] = kp;
    }
  }
}

extern "C" void kernel_launch(void* const* d_in, const int* in_sizes, int n_in,
                              void* d_out, int out_size, void* d_ws, size_t ws_size,
                              hipStream_t stream) {
  (void)in_sizes; (void)n_in; (void)out_size; (void)ws_size;
  const float* x   = (const float*)d_in[0];
  const float* h   = (const float*)d_in[1];
  const float* kpv = (const float*)d_in[2];
  const float* kpp = (const float*)d_in[3];
  const float* Wx  = (const float*)d_in[4];
  const float* bx  = (const float*)d_in[5];
  const float* Uh  = (const float*)d_in[6];
  const float* bh  = (const float*)d_in[7];

  unsigned short* Wpack = (unsigned short*)d_ws;   // 1280*512*2 = 1.31 MB
  float* out = (float*)d_out;

  pack_weights<<<(NCOL * KDIM) / 256, 256, 0, stream>>>(Wx, Uh, Wpack);

  dim3 grid(B_DIM / 16, 1, 1);   // 4096 M-tiles, full H per block
  hblstm_kernel<<<grid, 256, 0, stream>>>(x, h, kpv, kpp, bx, bh, Wpack, out);
}
